// GCN_55241869361249
// MI455X (gfx1250) — matrix operand order repaired and checked by reading, not compiled, not run on hardware
//
#include <hip/hip_runtime.h>
#include <hip/hip_bf16.h>

// ---------------------------------------------------------------------------
// GCN 2-layer forward for gfx1250 (MI455X): f16 WMMA GEMMs + L2-resident
// gather/scatter aggregation with f32 atomics.
// ---------------------------------------------------------------------------

typedef _Float16 half8 __attribute__((ext_vector_type(8)));
typedef _Float16 v16h  __attribute__((ext_vector_type(16)));
typedef float    v8f   __attribute__((ext_vector_type(8)));

#define GCN_N 100000
#define GCN_FIN 128
#define GCN_H 128
#define GCN_C 64

// ---------------- degree / normalization -----------------------------------

__global__ void gcn_deg_init(float* __restrict__ deg, int n) {
  int i = blockIdx.x * blockDim.x + threadIdx.x;
  if (i < n) deg[i] = 1.0f;  // self-loop contributes 1 to every node
}

__global__ void gcn_deg_count(const long long* __restrict__ row,
                              float* __restrict__ deg, int E) {
  int e = blockIdx.x * blockDim.x + threadIdx.x;
  if (e < E) atomicAdd(&deg[(int)row[e]], 1.0f);
}

__global__ void gcn_dinv(const float* __restrict__ deg,
                         float* __restrict__ dinv, int n) {
  int i = blockIdx.x * blockDim.x + threadIdx.x;
  if (i < n) dinv[i] = rsqrtf(deg[i]);  // deg >= 1 always (self-loops)
}

// ---------------- conversions ----------------------------------------------

__global__ void gcn_f32_to_f16(const float* __restrict__ src,
                               _Float16* __restrict__ dst, long total) {
  long i = (long)blockIdx.x * blockDim.x + threadIdx.x;
  if (i < total) dst[i] = (_Float16)src[i];
}

// W: [K,N] row-major f32  ->  Wt: [N,K] row-major f16 (column-major W)
__global__ void gcn_transpose_f16(const float* __restrict__ W,
                                  _Float16* __restrict__ Wt, int K, int N) {
  int idx = blockIdx.x * blockDim.x + threadIdx.x;
  if (idx < K * N) {
    int c = idx / K;
    int k = idx - c * K;
    Wt[idx] = (_Float16)W[k * N + c];
  }
}

// ---------------- WMMA GEMM: D[M,NDIM] = A[M,128] * B[128,NDIM] -------------
// A row-major f16 (stride 128). Bt = B^T row-major f16 ([NDIM,128]).
// One wave handles a 16-row strip and all NDIM/16 column tiles.
// Fragment layouts per CDNA5 ISA 7.12.2 (wave32):
//   A 16x32 f16: lane l -> row l&15; halves 0..7 = K k0..k0+7,
//                halves 8..15 = K k0+16..k0+23, k0 = 8*(l>>4)
//   B 32x16 f16: lane l -> col l&15; same K mapping as A
//   C/D 16x16 f32: lane l -> col l&15; vgpr v -> row v + 8*(l>>4)
template <int NDIM>
__global__ __launch_bounds__(64) void gcn_gemm_wmma(
    const _Float16* __restrict__ A, const _Float16* __restrict__ Bt,
    _Float16* __restrict__ D) {
  constexpr int K = 128;
  constexpr int NT = NDIM / 16;

  const int wave  = threadIdx.x >> 5;
  const int lane  = threadIdx.x & 31;
  const int strip = blockIdx.x * 2 + wave;  // 16-row strip; exact fit
  const int r  = lane & 15;
  const int hi = lane >> 4;

  const _Float16* arow = A + ((size_t)strip * 16 + r) * K;

  v8f acc[NT];
#pragma unroll
  for (int t = 0; t < NT; ++t) {
    v8f z = {0.f, 0.f, 0.f, 0.f, 0.f, 0.f, 0.f, 0.f};
    acc[t] = z;
  }

#pragma unroll
  for (int kk = 0; kk < K; kk += 32) {
    half8 alo = *(const half8*)(arow + kk + hi * 8);
    half8 ahi = *(const half8*)(arow + kk + hi * 8 + 16);
    v16h af = __builtin_shufflevector(alo, ahi, 0, 1, 2, 3, 4, 5, 6, 7, 8, 9,
                                      10, 11, 12, 13, 14, 15);
#pragma unroll
    for (int t = 0; t < NT; ++t) {
      const _Float16* bcol = Bt + ((size_t)t * 16 + r) * K + kk + hi * 8;
      half8 blo = *(const half8*)(bcol);
      half8 bhi = *(const half8*)(bcol + 16);
      v16h bf = __builtin_shufflevector(blo, bhi, 0, 1, 2, 3, 4, 5, 6, 7, 8, 9,
                                        10, 11, 12, 13, 14, 15);
      acc[t] = __builtin_amdgcn_wmma_f32_16x16x32_f16(
          false, af, false, bf, (short)0, acc[t], false, false);
    }
  }

#pragma unroll
  for (int t = 0; t < NT; ++t) {
    _Float16* drow = D + ((size_t)strip * 16 + hi * 8) * NDIM + t * 16 + r;
#pragma unroll
    for (int v = 0; v < 8; ++v) drow[(size_t)v * NDIM] = (_Float16)acc[t][v];
  }
}

// ---------------- aggregation ----------------------------------------------

// acc[i,:] = h[i,:] * dinv[i]^2   (the self-loop term; also zero-initializes)
template <int F>
__global__ void gcn_selfloop(const _Float16* __restrict__ h,
                             const float* __restrict__ dinv,
                             float* __restrict__ acc, long total) {
  long idx = (long)blockIdx.x * blockDim.x + threadIdx.x;
  if (idx >= total) return;
  int i = (int)(idx / F);
  float d = dinv[i];
  acc[idx] = (float)h[idx] * d * d;
}

// per-edge scatter-add: one thread per (edge, 8-feature chunk)
template <int F>
__global__ void gcn_scatter(const _Float16* __restrict__ h,
                            const long long* __restrict__ row,
                            const long long* __restrict__ col,
                            const float* __restrict__ dinv,
                            float* __restrict__ acc, int E) {
  constexpr int CH = F / 8;
  long idx = (long)blockIdx.x * blockDim.x + threadIdx.x;
  long e = idx / CH;
  if (e >= E) return;
  int c = (int)(idx - e * CH) * 8;
  int d = (int)row[e];
  int s = (int)col[e];
  float nrm = dinv[d] * dinv[s];
  half8 v = *(const half8*)(h + (size_t)s * F + c);
  float* a = acc + (size_t)d * F + c;
#pragma unroll
  for (int j = 0; j < 8; ++j) atomicAdd(a + j, (float)v[j] * nrm);
}

// hh = relu(acc + b1) cast to f16
__global__ void gcn_bias_relu_f16(const float* __restrict__ acc,
                                  const float* __restrict__ b,
                                  _Float16* __restrict__ out, long total) {
  long idx = (long)blockIdx.x * blockDim.x + threadIdx.x;
  if (idx >= total) return;
  int c = (int)(idx & (GCN_H - 1));
  out[idx] = (_Float16)fmaxf(acc[idx] + b[c], 0.0f);
}

// in-place: out[i,:] = log_softmax(out[i,:] + b2); one wave per node row (C=64)
__global__ void gcn_logsoftmax(float* __restrict__ out,
                               const float* __restrict__ b2, int n) {
  int node = blockIdx.x * (blockDim.x >> 5) + (threadIdx.x >> 5);
  if (node >= n) return;
  int lane = threadIdx.x & 31;
  float* p = out + (size_t)node * GCN_C;
  float z0 = p[lane] + b2[lane];
  float z1 = p[lane + 32] + b2[lane + 32];
  float m = fmaxf(z0, z1);
#pragma unroll
  for (int o = 16; o > 0; o >>= 1) m = fmaxf(m, __shfl_xor(m, o, 32));
  float s = expf(z0 - m) + expf(z1 - m);
#pragma unroll
  for (int o = 16; o > 0; o >>= 1) s += __shfl_xor(s, o, 32);
  float ls = logf(s);
  p[lane] = z0 - m - ls;
  p[lane + 32] = z1 - m - ls;
}

// ---------------------------------------------------------------------------

static inline size_t gcn_align(size_t x) { return (x + 255) & ~(size_t)255; }

extern "C" void kernel_launch(void* const* d_in, const int* in_sizes, int n_in,
                              void* d_out, int out_size, void* d_ws,
                              size_t ws_size, hipStream_t stream) {
  const float* x  = (const float*)d_in[0];
  const long long* ei = (const long long*)d_in[1];  // int64 [2, E]
  const float* W1 = (const float*)d_in[2];
  const float* b1 = (const float*)d_in[3];
  const float* W2 = (const float*)d_in[4];
  const float* b2 = (const float*)d_in[5];

  const int N = GCN_N;
  const int E = in_sizes[1] / 2;
  const long long* row = ei;      // destinations
  const long long* col = ei + E;  // sources

  // ---- workspace layout -------------------------------------------------
  char* w = (char*)d_ws;
  float* deg = (float*)w;            w += gcn_align((size_t)N * 4);
  float* dinv = (float*)w;           w += gcn_align((size_t)N * 4);
  _Float16* xh = (_Float16*)w;       w += gcn_align((size_t)N * GCN_FIN * 2);
  _Float16* w1t = (_Float16*)w;      w += gcn_align((size_t)GCN_H * GCN_FIN * 2);
  _Float16* w2t = (_Float16*)w;      w += gcn_align((size_t)GCN_C * GCN_H * 2);
  _Float16* h0h = (_Float16*)w;      w += gcn_align((size_t)N * GCN_H * 2);
  float* acc1 = (float*)w;           w += gcn_align((size_t)N * GCN_H * 4);
  _Float16* hh  = xh;                // alias: x(f16) dead after GEMM1
  _Float16* h1h = h0h;               // alias: h0 dead after scatter1
  float* acc2 = (float*)d_out;       // final buffer, finished in-place

  const int B = 256;

  // ---- normalization ----------------------------------------------------
  gcn_deg_init<<<(N + B - 1) / B, B, 0, stream>>>(deg, N);
  gcn_deg_count<<<(E + B - 1) / B, B, 0, stream>>>(row, deg, E);
  gcn_dinv<<<(N + B - 1) / B, B, 0, stream>>>(deg, dinv, N);

  // ---- conversions ------------------------------------------------------
  long xn = (long)N * GCN_FIN;
  gcn_f32_to_f16<<<(int)((xn + B - 1) / B), B, 0, stream>>>(x, xh, xn);
  gcn_transpose_f16<<<(GCN_FIN * GCN_H + B - 1) / B, B, 0, stream>>>(
      W1, w1t, GCN_FIN, GCN_H);
  gcn_transpose_f16<<<(GCN_H * GCN_C + B - 1) / B, B, 0, stream>>>(
      W2, w2t, GCN_H, GCN_C);

  // ---- layer 1 ----------------------------------------------------------
  gcn_gemm_wmma<GCN_H><<<N / 32, 64, 0, stream>>>(xh, w1t, h0h);

  long t1 = (long)N * GCN_H;
  gcn_selfloop<GCN_H><<<(int)((t1 + B - 1) / B), B, 0, stream>>>(h0h, dinv,
                                                                 acc1, t1);
  long s1 = (long)E * (GCN_H / 8);
  gcn_scatter<GCN_H><<<(int)((s1 + B - 1) / B), B, 0, stream>>>(
      h0h, row, col, dinv, acc1, E);
  gcn_bias_relu_f16<<<(int)((t1 + B - 1) / B), B, 0, stream>>>(acc1, b1, hh,
                                                               t1);

  // ---- layer 2 ----------------------------------------------------------
  gcn_gemm_wmma<GCN_C><<<N / 32, 64, 0, stream>>>(hh, w2t, h1h);

  long t2 = (long)N * GCN_C;
  gcn_selfloop<GCN_C><<<(int)((t2 + B - 1) / B), B, 0, stream>>>(h1h, dinv,
                                                                 acc2, t2);
  long s2 = (long)E * (GCN_C / 8);
  gcn_scatter<GCN_C><<<(int)((s2 + B - 1) / B), B, 0, stream>>>(
      h1h, row, col, dinv, acc2, E);

  // ---- log-softmax (in-place, adds b2) ----------------------------------
  gcn_logsoftmax<<<(N + 7) / 8, 256, 0, stream>>>(acc2, b2, N);
}
